// GCBlockP1_15745350107645
// MI455X (gfx1250) — matrix operand order repaired and checked by reading, not compiled, not run on hardware
//
#include <hip/hip_runtime.h>
#include <math.h>

// ---------------------------------------------------------------------------
// GCBlock fused kernel for MI455X (gfx1250, wave32, WMMA f16->f32)
//   out = p.at[idx_i].add( tanh( i1 @ ii_W + ii_b ) )
//   i1  = sum_b basis[:,b] * ( h @ W2_b ),  W2_b[k,c] = pi_W2[k, 8c+b]
//   h   = [p1[idx_i], p1[idx_j]] @ pi_W1 + pi_b1
//   p1  = tanh( p @ pp_W + pp_b )
// ---------------------------------------------------------------------------

typedef _Float16 half8  __attribute__((ext_vector_type(8)));
typedef _Float16 half16 __attribute__((ext_vector_type(16)));
typedef float    v8f    __attribute__((ext_vector_type(8)));

union AFrag { half16 v; half8 g[2]; };

#define WMMA_F16(A, B, Cacc) \
  __builtin_amdgcn_wmma_f32_16x16x32_f16(false, (A), false, (B), (short)0, (Cacc), false, false)

// ---------------------------------------------------------------------------
// Kernel 1: p1[n,c] = (f16) tanh(p @ pp_W + pp_b);  out = p  (output base)
// ---------------------------------------------------------------------------
__global__ __launch_bounds__(64) void node_pp_kernel(
    const float* __restrict__ p, const float* __restrict__ ppW,
    const float* __restrict__ ppb, _Float16* __restrict__ p1,
    float* __restrict__ out, int nnodes)
{
  __shared__ float prow[64];
  const int n = blockIdx.x;
  if (n >= nnodes) return;
  const int c = threadIdx.x;
  const float pv = p[n * 64 + c];
  prow[c] = pv;
  out[n * 64 + c] = pv;       // initialize output with p (scatter adds on top)
  __syncthreads();
  float acc = ppb[c];
  #pragma unroll 8
  for (int k = 0; k < 64; ++k) acc = fmaf(prow[k], ppW[k * 64 + c], acc);
  p1[n * 64 + c] = (_Float16)tanhf(acc);
}

// ---------------------------------------------------------------------------
// Kernel 2: fused edge MLP. One wave32 handles 16 edges per group.
// All per-group branches are wave-uniform -> EXEC all-1s at every WMMA.
// B fragments are software-pipelined one k-step ahead (wcur/wnxt) so their
// lifetimes overlap the WMMAs: forces distinct registers and partial
// s_wait_dscnt counts instead of a full DS drain per WMMA.
// ---------------------------------------------------------------------------
__global__ __launch_bounds__(256) void edge_kernel(
    const _Float16* __restrict__ p1,
    const int* __restrict__ idx_i, const int* __restrict__ idx_j,
    const float* __restrict__ basis,
    const float* __restrict__ piW1, const float* __restrict__ pib1,
    const float* __restrict__ piW2,
    const float* __restrict__ iiW,  const float* __restrict__ iib,
    float* __restrict__ out, int ngroups, int total_waves)
{
  extern __shared__ __align__(16) char smem[];
  _Float16* W1t  = (_Float16*)smem;          // [64 cols][128 k]   16 KB (W1^T)
  _Float16* W2t  = W1t + 64 * 128;           // [8 b][64 c][64 k]  64 KB (W2_b^T)
  _Float16* Wiit = W2t + 8 * 64 * 64;        // [64 c][64 k]        8 KB (iiW^T)
  float*    b1s  = (float*)(Wiit + 64 * 64); // [64]
  float*    bis  = b1s + 64;                 // [64]
  _Float16* hbuf = (_Float16*)(bis + 64);    // [8 waves][16][64]  16 KB staging

  const int tid = threadIdx.x;
  // ---- cooperative transpose+convert of weights into LDS (once per block) --
  for (int m = tid; m < 64 * 128; m += 256) {
    const int col = m >> 7, k = m & 127;
    W1t[m] = (_Float16)piW1[k * 64 + col];
  }
  for (int m = tid; m < 8 * 64 * 64; m += 256) {
    const int k = m & 63, c = (m >> 6) & 63, b = m >> 12;
    W2t[m] = (_Float16)piW2[k * 512 + c * 8 + b];
  }
  for (int m = tid; m < 64 * 64; m += 256) {
    const int k = m & 63, c = m >> 6;
    Wiit[m] = (_Float16)iiW[k * 64 + c];
  }
  if (tid < 64) { b1s[tid] = pib1[tid]; bis[tid] = iib[tid]; }
  __syncthreads();

  const int wave = tid >> 5;
  const int lane = tid & 31;
  const int row  = lane & 15;   // matrix row (A/M) and column (B/N, C/N)
  const int hi   = lane >> 4;   // half-wave selector in WMMA layouts
  _Float16* hb = hbuf + wave * (16 * 64);
  const int gw = blockIdx.x * 8 + wave;

  for (int g = gw; g < ngroups; g += total_waves) {
    const long e0 = (long)g << 4;
    const int  mi = idx_i[e0 + row];
    const int  mj = idx_j[e0 + row];
    const _Float16* rowi = p1 + (long)mi * 64;
    const _Float16* rowj = p1 + (long)mj * 64;

    // hoisted global loads (independent of all LDS traffic)
    const float* bp = basis + (e0 + row) * 8;
    const float4 bb0 = *(const float4*)bp;
    const float4 bb1 = *(const float4*)(bp + 4);
    int nodev[8];
    #pragma unroll
    for (int v = 0; v < 8; ++v) nodev[v] = idx_i[e0 + v + 8 * hi];

    // ======== stage 1: h[16,64] = pair[16,128] @ W1 + b1  (16 WMMA) ========
    v8f acc1[4] = {};
    AFrag wcur[4], wnxt[4];
    #pragma unroll
    for (int n = 0; n < 4; ++n) {                    // preload k-step 0
      const _Float16* wp = W1t + (n * 16 + row) * 128 + 16 * hi;
      wcur[n].g[0] = *(const half8*)wp;
      wcur[n].g[1] = *(const half8*)(wp + 8);
    }
    #pragma unroll
    for (int kk = 0; kk < 4; ++kk) {
      const int kb = kk * 32;
      AFrag a;
      const int k0 = kb + 16 * hi;       // A halves: K in [k0,k0+8)
      const int k1 = kb + 16 * hi + 8;   //           K in [k1,k1+8)
      const _Float16* s0 = (k0 < 64) ? (rowi + k0) : (rowj + (k0 - 64));
      const _Float16* s1 = (k1 < 64) ? (rowi + k1) : (rowj + (k1 - 64));
      a.g[0] = *(const half8*)s0;
      a.g[1] = *(const half8*)s1;
      if (kk < 3) {                                  // prefetch next k-step
        #pragma unroll
        for (int n = 0; n < 4; ++n) {
          const _Float16* wp = W1t + (n * 16 + row) * 128 + kb + 32 + 16 * hi;
          wnxt[n].g[0] = *(const half8*)wp;
          wnxt[n].g[1] = *(const half8*)(wp + 8);
        }
      }
      #pragma unroll
      for (int n = 0; n < 4; ++n) acc1[n] = WMMA_F16(a.v, wcur[n].v, acc1[n]);
      #pragma unroll
      for (int n = 0; n < 4; ++n) wcur[n] = wnxt[n]; // SSA rotate (no movs)
    }
    // C layout: element (vgpr v, lane) = (M = v+8*hi, N = n*16+row)
    #pragma unroll
    for (int n = 0; n < 4; ++n) {
      const float bias = b1s[n * 16 + row];
      #pragma unroll
      for (int v = 0; v < 8; ++v)
        hb[(v + 8 * hi) * 64 + n * 16 + row] = (_Float16)(acc1[n][v] + bias);
    }
    // preload stage-2 step 0 B fragments before the drain (W2t is read-only)
    #pragma unroll
    for (int n = 0; n < 4; ++n) {
      const _Float16* wp = W2t + (n * 16 + row) * 64 + 16 * hi;
      wcur[n].g[0] = *(const half8*)wp;
      wcur[n].g[1] = *(const half8*)(wp + 8);
    }
    asm volatile("s_wait_dscnt 0" ::: "memory");  // wave-local LDS ordering

    // ==== stage 2: i1 = sum_b (h*basis_b) @ W2_b  (64 WMMA, basis folded) ===
    const _Float16 bh[8] = {(_Float16)bb0.x, (_Float16)bb0.y, (_Float16)bb0.z,
                            (_Float16)bb0.w, (_Float16)bb1.x, (_Float16)bb1.y,
                            (_Float16)bb1.z, (_Float16)bb1.w};
    AFrag hc[2];                                  // h chunks, kept in registers
    #pragma unroll
    for (int kk = 0; kk < 2; ++kk) {
      const _Float16* hp = hb + row * 64 + kk * 32;
      hc[kk].g[0] = *(const half8*)(hp + 16 * hi);
      hc[kk].g[1] = *(const half8*)(hp + 16 * hi + 8);
    }
    v8f acc2[4] = {};
    #pragma unroll
    for (int t = 0; t < 16; ++t) {                // t = b*2 + kk pipeline
      const int b = t >> 1, kk = t & 1;
      if (t < 15) {                               // prefetch next step's B
        const int b2 = (t + 1) >> 1, kk2 = (t + 1) & 1;
        #pragma unroll
        for (int n = 0; n < 4; ++n) {
          const _Float16* wp =
              W2t + (b2 * 64 + n * 16 + row) * 64 + kk2 * 32 + 16 * hi;
          wnxt[n].g[0] = *(const half8*)wp;
          wnxt[n].g[1] = *(const half8*)(wp + 8);
        }
      }
      AFrag a;                                    // A rows scaled by basis[e,b]
      a.g[0] = hc[kk].g[0] * bh[b];
      a.g[1] = hc[kk].g[1] * bh[b];
      #pragma unroll
      for (int n = 0; n < 4; ++n) acc2[n] = WMMA_F16(a.v, wcur[n].v, acc2[n]);
      #pragma unroll
      for (int n = 0; n < 4; ++n) wcur[n] = wnxt[n];
    }
    #pragma unroll
    for (int n = 0; n < 4; ++n)
      #pragma unroll
      for (int v = 0; v < 8; ++v)
        hb[(v + 8 * hi) * 64 + n * 16 + row] = (_Float16)acc2[n][v];
    // preload stage-3 step 0 B fragments before the drain (Wiit read-only)
    #pragma unroll
    for (int n = 0; n < 4; ++n) {
      const _Float16* wp = Wiit + (n * 16 + row) * 64 + 16 * hi;
      wcur[n].g[0] = *(const half8*)wp;
      wcur[n].g[1] = *(const half8*)(wp + 8);
    }
    asm volatile("s_wait_dscnt 0" ::: "memory");

    // ======== stage 3: i2 = tanh(i1 @ iiW + iib); scatter  (8 WMMA) ========
    v8f acc3[4] = {};
    #pragma unroll
    for (int kk = 0; kk < 2; ++kk) {
      const int kb = kk * 32;
      AFrag a;
      const _Float16* hp = hb + row * 64 + kb;
      a.g[0] = *(const half8*)(hp + 16 * hi);
      a.g[1] = *(const half8*)(hp + 16 * hi + 8);
      if (kk == 0) {
        #pragma unroll
        for (int n = 0; n < 4; ++n) {
          const _Float16* wp = Wiit + (n * 16 + row) * 64 + 32 + 16 * hi;
          wnxt[n].g[0] = *(const half8*)wp;
          wnxt[n].g[1] = *(const half8*)(wp + 8);
        }
      }
      #pragma unroll
      for (int n = 0; n < 4; ++n) acc3[n] = WMMA_F16(a.v, wcur[n].v, acc3[n]);
      #pragma unroll
      for (int n = 0; n < 4; ++n) wcur[n] = wnxt[n];
    }
    #pragma unroll
    for (int v = 0; v < 8; ++v) {
      float* op = out + (long)nodev[v] * 64 + row;
      #pragma unroll
      for (int n = 0; n < 4; ++n)
        atomicAdd(op + n * 16, tanhf(acc3[n][v] + bis[n * 16 + row]));
    }
  }
}

// ---------------------------------------------------------------------------
// Scalar tail for E % 16 edges (not hit for E = 1,000,000)
// ---------------------------------------------------------------------------
__global__ __launch_bounds__(64) void edge_tail_kernel(
    const _Float16* __restrict__ p1, const int* __restrict__ idx_i,
    const int* __restrict__ idx_j, const float* __restrict__ basis,
    const float* __restrict__ piW1, const float* __restrict__ pib1,
    const float* __restrict__ piW2, const float* __restrict__ iiW,
    const float* __restrict__ iib, float* __restrict__ out, int estart, int E)
{
  const int e = estart + blockIdx.x;
  if (e >= E) return;
  __shared__ float pair[128];
  __shared__ float hsh[64];
  __shared__ float i1sh[64];
  const int c = threadIdx.x;
  const int i = idx_i[e], j = idx_j[e];
  pair[c]      = (float)p1[(long)i * 64 + c];
  pair[c + 64] = (float)p1[(long)j * 64 + c];
  __syncthreads();
  float h = pib1[c];
  for (int k = 0; k < 128; ++k) h = fmaf(pair[k], piW1[k * 64 + c], h);
  hsh[c] = h;
  __syncthreads();
  float i1 = 0.f;
  for (int b = 0; b < 8; ++b) {
    const float s = basis[(long)e * 8 + b];
    float t = 0.f;
    for (int k = 0; k < 64; ++k) t = fmaf(hsh[k], piW2[k * 512 + c * 8 + b], t);
    i1 = fmaf(s, t, i1);
  }
  i1sh[c] = i1;
  __syncthreads();
  float o = iib[c];
  for (int k = 0; k < 64; ++k) o = fmaf(i1sh[k], iiW[k * 64 + c], o);
  atomicAdd(&out[(long)i * 64 + c], tanhf(o));
}

// ---------------------------------------------------------------------------
extern "C" void kernel_launch(void* const* d_in, const int* in_sizes, int n_in,
                              void* d_out, int out_size, void* d_ws, size_t ws_size,
                              hipStream_t stream)
{
  const float* p     = (const float*)d_in[0];
  const int*   idx_i = (const int*)d_in[1];
  const int*   idx_j = (const int*)d_in[2];
  const float* basis = (const float*)d_in[3];
  const float* ppW   = (const float*)d_in[4];
  const float* ppb   = (const float*)d_in[5];
  const float* piW1  = (const float*)d_in[6];
  const float* pib1  = (const float*)d_in[7];
  const float* piW2  = (const float*)d_in[8];
  const float* iiW   = (const float*)d_in[9];
  const float* iib   = (const float*)d_in[10];
  float* out = (float*)d_out;

  const int N = in_sizes[0] / 64;
  const int E = in_sizes[1];
  _Float16* p1 = (_Float16*)d_ws;   // N*64 f16 (2.56 MB for N=20000)

  node_pp_kernel<<<N, 64, 0, stream>>>(p, ppW, ppb, p1, out, N);

  const int ngroups = E >> 4;
  if (ngroups > 0) {
    const int blocks = 1024;
    const int total_waves = blocks * 8;
    const size_t lds = (size_t)(64 * 128 + 8 * 64 * 64 + 64 * 64) * 2  // weights f16
                     + (size_t)(64 + 64) * 4                           // biases f32
                     + (size_t)8 * 16 * 64 * 2;                        // h/i1 staging
    edge_kernel<<<blocks, 256, lds, stream>>>(p1, idx_i, idx_j, basis,
        piW1, pib1, piW2, iiW, iib, out, ngroups, total_waves);
  }
  const int tail = E & 15;
  if (tail) {
    edge_tail_kernel<<<tail, 64, 0, stream>>>(p1, idx_i, idx_j, basis,
        piW1, pib1, piW2, iiW, iib, out, ngroups << 4, E);
  }
  (void)n_in; (void)out_size; (void)ws_size;
}
